// GlobalPool_29661044146426
// MI455X (gfx1250) — compile-verified, matrix operand order checked by reading, not run on hardware
//
#include <hip/hip_runtime.h>
#include <stdint.h>

#define V_N 500000
#define G_N 20000
#define F_N 128

typedef __attribute__((ext_vector_type(16))) __bf16        v16bf;
typedef __attribute__((ext_vector_type(8)))  __bf16        v8bf;
typedef __attribute__((ext_vector_type(8)))  float         v8f;
typedef __attribute__((ext_vector_type(8)))  unsigned int  v8u;

// ---- WMMA fragment loaders (16-bit A/B 16x32 layout, ISA 7.12.2) ----
// lane m = lane&15 selects row; hi = lane>>4 selects K-offset (+8);
// VGPR r in group g=r/4 holds K = g*16 + hi*8 + (r%4)*2 (+0,+1)
// => per lane: two contiguous 8-element chunks at k0+hi*8 and k0+16+hi*8.
static __device__ __forceinline__ v16bf load_frag_f32(const float* __restrict__ row,
                                                      int k0, int hi) {
    v8f f0 = *(const v8f*)(row + k0 + hi * 8);
    v8f f1 = *(const v8f*)(row + k0 + 16 + hi * 8);
    v8bf b0 = __builtin_convertvector(f0, v8bf);   // native packed f32->bf16 cvt
    v8bf b1 = __builtin_convertvector(f1, v8bf);
    return __builtin_shufflevector(b0, b1, 0, 1, 2, 3, 4, 5, 6, 7,
                                           8, 9, 10, 11, 12, 13, 14, 15);
}
static __device__ __forceinline__ v16bf load_frag_bf16(const __bf16* __restrict__ row,
                                                       int k0, int hi) {
    uint4 c0 = *(const uint4*)(row + k0 + hi * 8);
    uint4 c1 = *(const uint4*)(row + k0 + 16 + hi * 8);
    v8u u;
    u[0] = c0.x; u[1] = c0.y; u[2] = c0.z; u[3] = c0.w;
    u[4] = c1.x; u[5] = c1.y; u[6] = c1.z; u[7] = c1.w;
    return __builtin_bit_cast(v16bf, u);
}
static __device__ __forceinline__ v8f wmma_bf16(v16bf a, v16bf b, v8f c) {
    return __builtin_amdgcn_wmma_f32_16x16x32_bf16(false, a, false, b, (short)0, c, false, false);
}

// ---- 1) weights -> bf16 scratch copies ----
__global__ __launch_bounds__(256) void k_prep(const float* __restrict__ wproj,
                                              const float* __restrict__ wih,
                                              const float* __restrict__ whh,
                                              __bf16* __restrict__ wprojb,
                                              __bf16* __restrict__ wihb,
                                              __bf16* __restrict__ whhb) {
    int i = blockIdx.x * 256 + threadIdx.x;
    if (i < F_N * F_N) wprojb[i] = (__bf16)wproj[i];
    if (i < 3 * F_N * F_N) { wihb[i] = (__bf16)wih[i]; whhb[i] = (__bf16)whh[i]; }
}

// ---- 2) per-graph: dot(relu(g_feats[g]), W_logit[0:F]) ----
__global__ __launch_bounds__(256) void k_glogit(const float* __restrict__ gfeat,
                                                const float* __restrict__ wlog,
                                                float* __restrict__ glogit) {
    int w = threadIdx.x >> 5, lane = threadIdx.x & 31;
    int g = blockIdx.x * 8 + w;
    if (g >= G_N) return;
    float4 x = ((const float4*)(gfeat + (size_t)g * F_N))[lane];
    float4 wv = ((const float4*)wlog)[lane];
    float s = fmaxf(x.x, 0.f) * wv.x + fmaxf(x.y, 0.f) * wv.y +
              fmaxf(x.z, 0.f) * wv.z + fmaxf(x.w, 0.f) * wv.w;
    for (int off = 16; off >= 1; off >>= 1) s += __shfl_xor(s, off, 32);
    if (lane == 0) glogit[g] = s;
}

// ---- 3) per-node logit + LeakyReLU -> z ----
__global__ __launch_bounds__(256) void k_zlogit(const float* __restrict__ node,
                                                const int* __restrict__ seg,
                                                const float* __restrict__ glogit,
                                                const float* __restrict__ wlog,
                                                const float* __restrict__ blog,
                                                float* __restrict__ z) {
    __shared__ float s_wl[F_N];
    if (threadIdx.x < F_N) s_wl[threadIdx.x] = wlog[F_N + threadIdx.x];
    __syncthreads();
    int t = threadIdx.x;
    int v = blockIdx.x * 64 + (t >> 2);
    int q = t & 3;
    float s = 0.f;
    if (v < V_N) {
        const float4* row = (const float4*)(node + (size_t)v * F_N + q * 32);
        const float4* wl  = (const float4*)(s_wl + q * 32);
        #pragma unroll
        for (int i = 0; i < 8; ++i) {
            float4 x = row[i], ww = wl[i];
            s += x.x * ww.x + x.y * ww.y + x.z * ww.z + x.w * ww.w;
        }
    }
    s += __shfl_xor(s, 1, 32);
    s += __shfl_xor(s, 2, 32);
    if (v < V_N && q == 0) {
        float val = s + glogit[seg[v]] + blog[0];
        z[v] = val > 0.f ? val : 0.01f * val;
    }
}

// ---- 4) CSR offsets from sorted seg ----
__global__ __launch_bounds__(256) void k_offsets(const int* __restrict__ seg,
                                                 int* __restrict__ offs) {
    int v = blockIdx.x * 256 + threadIdx.x;
    if (v >= V_N) return;
    int s  = seg[v];
    int sp = (v == 0) ? -1 : seg[v - 1];
    for (int g = sp + 1; g <= s; ++g) offs[g] = v;
    if (v == V_N - 1)
        for (int g = s + 1; g <= G_N; ++g) offs[g] = V_N;
}

// ---- 5) one wave per graph: segment softmax -> att ----
__global__ __launch_bounds__(256) void k_segsoftmax(const float* __restrict__ z,
                                                    const int* __restrict__ offs,
                                                    float* __restrict__ att) {
    int w = threadIdx.x >> 5, lane = threadIdx.x & 31;
    int g = blockIdx.x * 8 + w;
    if (g >= G_N) return;
    int s0 = offs[g], s1 = offs[g + 1];
    float m = -3.4e38f;
    for (int v = s0 + lane; v < s1; v += 32) m = fmaxf(m, z[v]);
    for (int off = 16; off >= 1; off >>= 1) m = fmaxf(m, __shfl_xor(m, off, 32));
    float sum = 0.f;
    for (int v = s0 + lane; v < s1; v += 32) sum += __expf(z[v] - m);
    for (int off = 16; off >= 1; off >>= 1) sum += __shfl_xor(sum, off, 32);
    float inv = (sum > 0.f) ? 1.f / sum : 0.f;
    for (int v = s0 + lane; v < s1; v += 32) att[v] = __expf(z[v] - m) * inv;
}

// ---- 6) main WMMA GEMM: hv = node @ Wproj^T, weighted segment-sum ----
__global__ __launch_bounds__(256) void k_pool_gemm(const float* __restrict__ node,
                                                   const int* __restrict__ seg,
                                                   const float* __restrict__ att,
                                                   const __bf16* __restrict__ wprojb,
                                                   float* __restrict__ g_repr) {
    __shared__ float s_tile[128 * 132];
    __shared__ float s_att[128];
    __shared__ int   s_seg[128];
    const int t = threadIdx.x;
    const int base = blockIdx.x * 128;
    if (t < 128) {
        int v = base + t;
        if (v < V_N) { s_att[t] = att[v]; s_seg[t] = seg[v]; }
        else         { s_att[t] = 0.f;    s_seg[t] = seg[V_N - 1]; }
    }
    __syncthreads();
    const int lane = t & 31, w = t >> 5;
    const int m = lane & 15, hi = lane >> 4;
    int row = base + w * 16 + m;
    if (row >= V_N) row = V_N - 1;           // att=0 for OOB rows kills contribution
    const float* arow = node + (size_t)row * F_N;
    v16bf a0 = load_frag_f32(arow,  0, hi);
    v16bf a1 = load_frag_f32(arow, 32, hi);
    v16bf a2 = load_frag_f32(arow, 64, hi);
    v16bf a3 = load_frag_f32(arow, 96, hi);
    for (int nt = 0; nt < 8; ++nt) {
        const __bf16* brow = wprojb + (size_t)(nt * 16 + m) * F_N;
        v8f acc = {};
        acc = wmma_bf16(a0, load_frag_bf16(brow,  0, hi), acc);
        acc = wmma_bf16(a1, load_frag_bf16(brow, 32, hi), acc);
        acc = wmma_bf16(a2, load_frag_bf16(brow, 64, hi), acc);
        acc = wmma_bf16(a3, load_frag_bf16(brow, 96, hi), acc);
        #pragma unroll
        for (int i = 0; i < 8; ++i) {
            int rl = w * 16 + hi * 8 + i;    // D layout: VGPR i -> row i + 8*hi, col = lane&15
            s_tile[rl * 132 + nt * 16 + m] = acc[i] * s_att[rl];
        }
    }
    __syncthreads();
    if (t < 128) {                            // per-column run-length segment reduction
        int col = t;
        float sum = 0.f;
        int cur = s_seg[0];
        for (int r = 0; r < 128; ++r) {
            int sg = s_seg[r];
            if (sg != cur) {
                atomicAdd(&g_repr[(size_t)cur * F_N + col], sum);
                sum = 0.f; cur = sg;
            }
            sum += s_tile[r * 132 + col];
        }
        atomicAdd(&g_repr[(size_t)cur * F_N + col], sum);
    }
}

// ---- 7) context = elu(g_repr + b_proj) -> bf16 ----
__global__ __launch_bounds__(256) void k_context(const float* __restrict__ g_repr,
                                                 const float* __restrict__ bproj,
                                                 __bf16* __restrict__ ctxb) {
    int i = blockIdx.x * 256 + threadIdx.x;
    if (i >= G_N * F_N) return;
    float v = g_repr[i] + bproj[i & (F_N - 1)];
    v = v > 0.f ? v : (__expf(v) - 1.f);
    ctxb[i] = (__bf16)v;
}

// ---- 8) GRU: gi = ctx@w_ih^T, gh = h@w_hh^T, gates fused ----
__global__ __launch_bounds__(256) void k_gru(const __bf16* __restrict__ ctxb,
                                             const float* __restrict__ gfeat,
                                             const __bf16* __restrict__ wihb,
                                             const __bf16* __restrict__ whhb,
                                             const float* __restrict__ bih,
                                             const float* __restrict__ bhh,
                                             float* __restrict__ out) {
    const int t = threadIdx.x;
    const int lane = t & 31, w = t >> 5;
    const int m = lane & 15, hi = lane >> 4;
    const int base = blockIdx.x * 128;
    int row = base + w * 16 + m;
    int rc = row < G_N ? row : G_N - 1;
    const __bf16* crow = ctxb + (size_t)rc * F_N;
    const float*  hrow = gfeat + (size_t)rc * F_N;
    v16bf c0 = load_frag_bf16(crow,  0, hi), c1 = load_frag_bf16(crow, 32, hi);
    v16bf c2 = load_frag_bf16(crow, 64, hi), c3 = load_frag_bf16(crow, 96, hi);
    v16bf h0 = load_frag_f32(hrow,  0, hi), h1 = load_frag_f32(hrow, 32, hi);
    v16bf h2 = load_frag_f32(hrow, 64, hi), h3 = load_frag_f32(hrow, 96, hi);
    for (int j = 0; j < 8; ++j) {
        const __bf16* wr_i = wihb + (size_t)(          j * 16 + m) * F_N;
        const __bf16* wz_i = wihb + (size_t)(    F_N + j * 16 + m) * F_N;
        const __bf16* wn_i = wihb + (size_t)(2 * F_N + j * 16 + m) * F_N;
        const __bf16* wr_h = whhb + (size_t)(          j * 16 + m) * F_N;
        const __bf16* wz_h = whhb + (size_t)(    F_N + j * 16 + m) * F_N;
        const __bf16* wn_h = whhb + (size_t)(2 * F_N + j * 16 + m) * F_N;
        v8f ir = {}, iz = {}, in_ = {}, hr = {}, hz = {}, hn = {};
        ir  = wmma_bf16(c0, load_frag_bf16(wr_i,  0, hi), ir);
        ir  = wmma_bf16(c1, load_frag_bf16(wr_i, 32, hi), ir);
        ir  = wmma_bf16(c2, load_frag_bf16(wr_i, 64, hi), ir);
        ir  = wmma_bf16(c3, load_frag_bf16(wr_i, 96, hi), ir);
        iz  = wmma_bf16(c0, load_frag_bf16(wz_i,  0, hi), iz);
        iz  = wmma_bf16(c1, load_frag_bf16(wz_i, 32, hi), iz);
        iz  = wmma_bf16(c2, load_frag_bf16(wz_i, 64, hi), iz);
        iz  = wmma_bf16(c3, load_frag_bf16(wz_i, 96, hi), iz);
        in_ = wmma_bf16(c0, load_frag_bf16(wn_i,  0, hi), in_);
        in_ = wmma_bf16(c1, load_frag_bf16(wn_i, 32, hi), in_);
        in_ = wmma_bf16(c2, load_frag_bf16(wn_i, 64, hi), in_);
        in_ = wmma_bf16(c3, load_frag_bf16(wn_i, 96, hi), in_);
        hr  = wmma_bf16(h0, load_frag_bf16(wr_h,  0, hi), hr);
        hr  = wmma_bf16(h1, load_frag_bf16(wr_h, 32, hi), hr);
        hr  = wmma_bf16(h2, load_frag_bf16(wr_h, 64, hi), hr);
        hr  = wmma_bf16(h3, load_frag_bf16(wr_h, 96, hi), hr);
        hz  = wmma_bf16(h0, load_frag_bf16(wz_h,  0, hi), hz);
        hz  = wmma_bf16(h1, load_frag_bf16(wz_h, 32, hi), hz);
        hz  = wmma_bf16(h2, load_frag_bf16(wz_h, 64, hi), hz);
        hz  = wmma_bf16(h3, load_frag_bf16(wz_h, 96, hi), hz);
        hn  = wmma_bf16(h0, load_frag_bf16(wn_h,  0, hi), hn);
        hn  = wmma_bf16(h1, load_frag_bf16(wn_h, 32, hi), hn);
        hn  = wmma_bf16(h2, load_frag_bf16(wn_h, 64, hi), hn);
        hn  = wmma_bf16(h3, load_frag_bf16(wn_h, 96, hi), hn);
        int ncol = j * 16 + m;
        float bir = bih[ncol], biz = bih[F_N + ncol], bin = bih[2 * F_N + ncol];
        float bhr = bhh[ncol], bhz = bhh[F_N + ncol], bhn = bhh[2 * F_N + ncol];
        #pragma unroll
        for (int i = 0; i < 8; ++i) {
            int g = base + w * 16 + hi * 8 + i;
            if (g < G_N) {
                float h  = gfeat[(size_t)g * F_N + ncol];
                float r  = 1.f / (1.f + __expf(-(ir[i] + bir + hr[i] + bhr)));
                float u  = 1.f / (1.f + __expf(-(iz[i] + biz + hz[i] + bhz)));
                float nn = tanhf(in_[i] + bin + r * (hn[i] + bhn));
                out[(size_t)g * F_N + ncol] = (1.f - u) * nn + u * h;
            }
        }
    }
}

extern "C" void kernel_launch(void* const* d_in, const int* in_sizes, int n_in,
                              void* d_out, int out_size, void* d_ws, size_t ws_size,
                              hipStream_t stream) {
    const float* node  = (const float*)d_in[0];
    const float* gfeat = (const float*)d_in[1];
    const int*   seg   = (const int*)d_in[2];
    const float* wlog  = (const float*)d_in[3];
    const float* blog  = (const float*)d_in[4];
    const float* wproj = (const float*)d_in[5];
    const float* bproj = (const float*)d_in[6];
    const float* wih   = (const float*)d_in[7];
    const float* whh   = (const float*)d_in[8];
    const float* bih   = (const float*)d_in[9];
    const float* bhh   = (const float*)d_in[10];
    float* out = (float*)d_out;

    char* ws = (char*)d_ws;
    size_t o = 0;
    auto alloc = [&](size_t bytes) -> void* {
        void* p = ws + o;
        o += (bytes + 255) & ~(size_t)255;
        return p;
    };
    float* g_repr   = (float*)alloc((size_t)G_N * F_N * 4);
    float* z        = (float*)alloc((size_t)V_N * 4);
    float* att      = (float*)alloc((size_t)V_N * 4);
    float* glogit   = (float*)alloc((size_t)G_N * 4);
    int*   offs     = (int*)alloc((size_t)(G_N + 1) * 4);
    __bf16* ctxb    = (__bf16*)alloc((size_t)G_N * F_N * 2);
    __bf16* wprojb  = (__bf16*)alloc((size_t)F_N * F_N * 2);
    __bf16* wihb    = (__bf16*)alloc((size_t)3 * F_N * F_N * 2);
    __bf16* whhb    = (__bf16*)alloc((size_t)3 * F_N * F_N * 2);

    hipMemsetAsync(g_repr, 0, (size_t)G_N * F_N * 4, stream);
    k_prep<<<(3 * F_N * F_N + 255) / 256, 256, 0, stream>>>(wproj, wih, whh, wprojb, wihb, whhb);
    k_glogit<<<(G_N + 7) / 8, 256, 0, stream>>>(gfeat, wlog, glogit);
    k_zlogit<<<(V_N + 63) / 64, 256, 0, stream>>>(node, seg, glogit, wlog, blog, z);
    k_offsets<<<(V_N + 255) / 256, 256, 0, stream>>>(seg, offs);
    k_segsoftmax<<<(G_N + 7) / 8, 256, 0, stream>>>(z, offs, att);
    k_pool_gemm<<<(V_N + 127) / 128, 256, 0, stream>>>(node, seg, att, wprojb, g_repr);
    k_context<<<(G_N * F_N + 255) / 256, 256, 0, stream>>>(g_repr, bproj, ctxb);
    k_gru<<<(G_N + 127) / 128, 256, 0, stream>>>(ctxb, gfeat, wihb, whhb, bih, bhh, out);
}
